// Block_15058155340426
// MI455X (gfx1250) — matrix operand order repaired and checked by reading, not compiled, not run on hardware
//
#include <hip/hip_runtime.h>
#include <hip/hip_bf16.h>

#define DIM 768
#define HEADS 12
#define HEAD_DIM 64
#define HIDDEN 3072
#define SEQ 2048
#define BATCH 4
#define NTOK (BATCH * SEQ)   // 8192

typedef __bf16 bf16_t;
typedef __attribute__((ext_vector_type(16))) __bf16 v16bf;
typedef __attribute__((ext_vector_type(8)))  float  v8f;

__device__ __forceinline__ bf16_t f2bf(float f) { return static_cast<bf16_t>(f); }

__device__ __forceinline__ v8f wmma_bf16(v16bf a, v16bf b, v8f c) {
  // v_wmma_f32_16x16x32_bf16: (neg_a, A, neg_b, B, c_mod, C, reuse_a, reuse_b)
  return __builtin_amdgcn_wmma_f32_16x16x32_bf16(false, a, false, b, (short)0, c,
                                                 false, false);
}

// ---- gfx1250 async copy global -> LDS (ASYNCcnt), 16B per lane ----
__device__ __forceinline__ void async_copy_b128(unsigned lds_off, const void* gaddr) {
  asm volatile("global_load_async_to_lds_b128 %0, %1, off"
               :: "v"(lds_off), "v"(gaddr) : "memory");
}
__device__ __forceinline__ void wait_async_le6() {
  asm volatile("s_wait_asynccnt 0x6" ::: "memory");
}
__device__ __forceinline__ void wait_async_0() {
  asm volatile("s_wait_asynccnt 0x0" ::: "memory");
}

// ---- A-fragment (16x32 bf16, M x K): lane = half*16 + (m%16)
// ISA 7.12.2: v0..3 -> K = 8*half + 2v,2v+1 ; v4..7 -> K = 16 + 8*half + ...
__device__ __forceinline__ v16bf load_a_frag(const bf16_t* __restrict__ A, int lda,
                                             int mbase, int kbase, int lane) {
  const int half = lane >> 4;
  const int m = mbase + (lane & 15);
  const bf16_t* row = A + (size_t)m * lda + kbase;
  v16bf f;
#pragma unroll
  for (int v = 0; v < 4; ++v) {
    const int k0 = 8 * half + 2 * v;
    f[2 * v]     = row[k0];
    f[2 * v + 1] = row[k0 + 1];
    const int k1 = 16 + 8 * half + 2 * v;
    f[8 + 2 * v]     = row[k1];
    f[8 + 2 * v + 1] = row[k1 + 1];
  }
  return f;
}

// ---- B-fragment (32x16 bf16, K x N) loaded from B^T storage [N][K]:
// elements 0..7 hold K = 8*half + 0..7, elements 8..15 hold K = 16 + 8*half + 0..7.
__device__ __forceinline__ v16bf load_bT_frag(const bf16_t* __restrict__ BT, int ldb,
                                              int nbase, int kbase, int lane) {
  const int half = lane >> 4;
  const int n = nbase + (lane & 15);
  const bf16_t* col = BT + (size_t)n * ldb + kbase + 8 * half;
  v16bf f;
#pragma unroll
  for (int v = 0; v < 8; ++v) {
    f[v]     = col[v];
    f[8 + v] = col[16 + v];
  }
  return f;
}

__device__ __forceinline__ float half16_max(float v) {
  v = fmaxf(v, __shfl_xor(v, 1, 32));
  v = fmaxf(v, __shfl_xor(v, 2, 32));
  v = fmaxf(v, __shfl_xor(v, 4, 32));
  v = fmaxf(v, __shfl_xor(v, 8, 32));
  return v;
}
__device__ __forceinline__ float half16_sum(float v) {
  v += __shfl_xor(v, 1, 32);
  v += __shfl_xor(v, 2, 32);
  v += __shfl_xor(v, 4, 32);
  v += __shfl_xor(v, 8, 32);
  return v;
}

// ======================= weight convert + transpose =======================
// in: f32 [K][N] row-major  ->  out: bf16 [N][K]
__global__ void convT_kernel(const float* __restrict__ w, bf16_t* __restrict__ wT,
                             int K, int N) {
  size_t i = (size_t)blockIdx.x * 256 + threadIdx.x;
  if (i >= (size_t)K * N) return;
  int k = (int)(i / N);
  int n = (int)(i % N);
  wT[(size_t)n * K + k] = f2bf(w[i]);
}

// ======================= LayerNorm -> bf16 =======================
__global__ void ln_kernel(const float* __restrict__ x, const float* __restrict__ g,
                          const float* __restrict__ b, bf16_t* __restrict__ out) {
  const int row = blockIdx.x;
  const float* xr = x + (size_t)row * DIM;
  __shared__ float red[256];
  float s = 0.f, s2 = 0.f;
  for (int c = threadIdx.x; c < DIM; c += 256) {
    float v = xr[c];
    s += v; s2 += v * v;
  }
  red[threadIdx.x] = s; __syncthreads();
  for (int off = 128; off > 0; off >>= 1) {
    if (threadIdx.x < off) red[threadIdx.x] += red[threadIdx.x + off];
    __syncthreads();
  }
  const float mean = red[0] / DIM;
  __syncthreads();
  red[threadIdx.x] = s2; __syncthreads();
  for (int off = 128; off > 0; off >>= 1) {
    if (threadIdx.x < off) red[threadIdx.x] += red[threadIdx.x + off];
    __syncthreads();
  }
  const float var = red[0] / DIM - mean * mean;
  const float inv = rsqrtf(var + 1e-5f);
  for (int c = threadIdx.x; c < DIM; c += 256) {
    float v = (xr[c] - mean) * inv * g[c] + b[c];
    out[(size_t)row * DIM + c] = f2bf(v);
  }
}

// ======================= WMMA GEMM, async-LDS double-buffered =======================
// C[M,N] = A[M,K](bf16) * B[K,N] (stored as BT[N][K] bf16)
// block = 256 threads (8 waves, 4x2), block tile 128x64, wave tile 32x32, K-step 64.
// Tiles staged via global_load_async_to_lds_b128 (6 async ops per wave per stage),
// double-buffered: "s_wait_asynccnt 6" drains stage kt while kt+1 stays in flight.
// 8 WMMAs per wave between barriers.
template <bool HAS_BIAS, bool HAS_RES, bool GELU_ACT, bool OUT_BF16>
__global__ __launch_bounds__(256, 1)
void gemm_wmma(const bf16_t* __restrict__ A, const bf16_t* __restrict__ BT,
               const float* __restrict__ bias, const float* __restrict__ res,
               void* __restrict__ outp, int M, int N, int K) {
  __shared__ bf16_t As[2][128 * 64];  // [buf][m][k]  16 KB each
  __shared__ bf16_t Bs[2][64 * 64];   // [buf][n][k]   8 KB each

  const int lane = threadIdx.x & 31;
  const int wave = threadIdx.x >> 5;
  const int wm = wave >> 1;           // 0..3
  const int wn = wave & 1;            // 0..1
  const int mblk = blockIdx.y * 128;
  const int nblk = blockIdx.x * 64;

  // stage one 128x64 A tile + 64x64 B^T tile into LDS buffer `buf`
  auto stage = [&](int kk, int buf) {
    // A: 1024 x 16B chunks -> 4 per thread (4 async instr per wave)
#pragma unroll
    for (int t = 0; t < 4; ++t) {
      const int c = threadIdx.x + t * 256;
      const int row = c >> 3, seg = c & 7;            // seg = 8 bf16 = 16 B
      const bf16_t* src = A + (size_t)(mblk + row) * K + kk + seg * 8;
      async_copy_b128((unsigned)(size_t)&As[buf][row * 64 + seg * 8], src);
    }
    // B: 512 x 16B chunks -> 2 per thread (2 async instr per wave)
#pragma unroll
    for (int t = 0; t < 2; ++t) {
      const int c = threadIdx.x + t * 256;
      const int row = c >> 3, seg = c & 7;
      const bf16_t* src = BT + (size_t)(nblk + row) * K + kk + seg * 8;
      async_copy_b128((unsigned)(size_t)&Bs[buf][row * 64 + seg * 8], src);
    }
  };

  const int T = K / 64;
  stage(0, 0);
  if (T > 1) stage(64, 1);

  v8f acc[2][2] = {};
  for (int kt = 0; kt < T; ++kt) {
    const int buf = kt & 1;
    if (kt + 1 < T) wait_async_le6();   // stage kt drained, stage kt+1 in flight
    else            wait_async_0();
    __syncthreads();

#pragma unroll
    for (int ks = 0; ks < 2; ++ks) {    // two 32-deep WMMA K-slices per stage
      v16bf a0 = load_a_frag(As[buf], 64, wm * 32, ks * 32, lane);
      v16bf a1 = load_a_frag(As[buf], 64, wm * 32 + 16, ks * 32, lane);
      v16bf b0 = load_bT_frag(Bs[buf], 64, wn * 32, ks * 32, lane);
      v16bf b1 = load_bT_frag(Bs[buf], 64, wn * 32 + 16, ks * 32, lane);
      acc[0][0] = wmma_bf16(a0, b0, acc[0][0]);
      acc[0][1] = wmma_bf16(a0, b1, acc[0][1]);
      acc[1][0] = wmma_bf16(a1, b0, acc[1][0]);
      acc[1][1] = wmma_bf16(a1, b1, acc[1][1]);
    }

    __syncthreads();                    // everyone done reading buf
    if (kt + 2 < T) stage((kt + 2) * 64, buf);
  }

  const int half = lane >> 4;
  const int nlane = lane & 15;
#pragma unroll
  for (int tm = 0; tm < 2; ++tm) {
#pragma unroll
    for (int tn = 0; tn < 2; ++tn) {
#pragma unroll
      for (int i = 0; i < 8; ++i) {
        const int m = mblk + wm * 32 + tm * 16 + i + 8 * half;
        const int n = nblk + wn * 32 + tn * 16 + nlane;
        float v = acc[tm][tn][i];
        if (HAS_BIAS) v += bias[n];
        const size_t idx = (size_t)m * N + n;
        if (HAS_RES) v += res[idx];
        if (GELU_ACT) v = 0.5f * v * (1.0f + erff(v * 0.70710678118654752f));
        if (OUT_BF16) ((bf16_t*)outp)[idx] = f2bf(v);
        else          ((float*)outp)[idx] = v;
      }
    }
  }
}

// ======================= flash attention (WMMA) =======================
// qkv: bf16 [NTOK][3*DIM]; o: bf16 [NTOK][DIM]
// block = 256 threads (8 waves); wave owns 16 query rows; key chunks of 32.
// K chunk staged to LDS via async b128 (1 per thread); V chunk staged transposed.
__global__ __launch_bounds__(256, 1)
void attn_kernel(const bf16_t* __restrict__ qkv, bf16_t* __restrict__ o) {
  const int b = blockIdx.z, h = blockIdx.y;
  const int lane = threadIdx.x & 31;
  const int wave = threadIdx.x >> 5;
  const int qbase = blockIdx.x * 128 + wave * 16;
  const int half = lane >> 4;
  const int nlane = lane & 15;
  const int rs = 3 * DIM;  // 2304

  const bf16_t* base = qkv + (size_t)b * SEQ * rs;
  const bf16_t* Qb = base + h * HEAD_DIM;            // + row*rs + d
  const bf16_t* Kb = base + DIM + h * HEAD_DIM;
  const bf16_t* Vb = base + 2 * DIM + h * HEAD_DIM;

  __shared__ bf16_t Ks[32 * HEAD_DIM];       // K chunk row-major: [key][dim] 4 KB
  __shared__ bf16_t vT[HEAD_DIM * 32];       // V chunk transposed: [dim][key] 4 KB
  __shared__ bf16_t pbuf[8][16 * 32];        // per-wave P tile 8 KB

  float mrow[8], lrow[8];
  v8f acc[4] = {};
#pragma unroll
  for (int i = 0; i < 8; ++i) { mrow[i] = -1e30f; lrow[i] = 0.f; }

  // Q fragments are loop-invariant: hd-K steps 0..31 and 32..63
  const v16bf qf0 = load_a_frag(Qb, rs, qbase, 0, lane);
  const v16bf qf1 = load_a_frag(Qb, rs, qbase, 32, lane);
  const float scale = 0.125f;  // 1/sqrt(64)

  for (int j = 0; j < SEQ; j += 32) {
    __syncthreads();
    // async-stage K chunk [32 keys][64 dims]: 256 x 16B chunks, 1 per thread
    {
      const int row = threadIdx.x >> 3, seg = threadIdx.x & 7;
      const bf16_t* src = Kb + (size_t)(j + row) * rs + seg * 8;
      async_copy_b128((unsigned)(size_t)&Ks[row * 64 + seg * 8], src);
    }
    // stage V^T chunk cooperatively: 64 dims x 32 keys (transpose, plain stores)
    for (int e = threadIdx.x; e < HEAD_DIM * 32; e += 256) {
      const int d = e & 63;
      const int kk = e >> 6;
      vT[d * 32 + kk] = Vb[(size_t)(j + kk) * rs + d];
    }
    wait_async_0();
    __syncthreads();

    // S = Q @ K^T for 32 keys (two 16-wide n-tiles, two K-steps over head dim)
    v8f s0 = {}, s1 = {};
    {
      v16bf b00 = load_bT_frag(Ks, 64, 0, 0, lane);
      v16bf b01 = load_bT_frag(Ks, 64, 0, 32, lane);
      s0 = wmma_bf16(qf0, b00, s0);
      s0 = wmma_bf16(qf1, b01, s0);
      v16bf b10 = load_bT_frag(Ks, 64, 16, 0, lane);
      v16bf b11 = load_bT_frag(Ks, 64, 16, 32, lane);
      s1 = wmma_bf16(qf0, b10, s1);
      s1 = wmma_bf16(qf1, b11, s1);
    }

    // online softmax update (rows i+8*half live within one 16-lane half)
#pragma unroll
    for (int i = 0; i < 8; ++i) {
      const float v0 = s0[i] * scale;
      const float v1 = s1[i] * scale;
      const float rmax = half16_max(fmaxf(v0, v1));
      const float mnew = fmaxf(mrow[i], rmax);
      const float corr = expf(mrow[i] - mnew);
      const float p0 = expf(v0 - mnew);
      const float p1 = expf(v1 - mnew);
      const float rsum = half16_sum(p0 + p1);
      lrow[i] = lrow[i] * corr + rsum;
      mrow[i] = mnew;
#pragma unroll
      for (int t = 0; t < 4; ++t) acc[t][i] *= corr;
      const int row = i + 8 * half;
      pbuf[wave][row * 32 + nlane]      = f2bf(p0);
      pbuf[wave][row * 32 + 16 + nlane] = f2bf(p1);
    }

    // O += P @ V  (A = P 16x32 via LDS, B = V^T fragments from LDS)
    const v16bf pf = load_a_frag(&pbuf[wave][0], 32, 0, 0, lane);
#pragma unroll
    for (int t = 0; t < 4; ++t) {
      v16bf vf = load_bT_frag(vT, 32, t * 16, 0, lane);
      acc[t] = wmma_bf16(pf, vf, acc[t]);
    }
  }

  // normalize and write O (bf16) at [token][h*64+d]
#pragma unroll
  for (int t = 0; t < 4; ++t) {
#pragma unroll
    for (int i = 0; i < 8; ++i) {
      const int row = qbase + i + 8 * half;
      const int d = t * 16 + nlane;
      const float v = acc[t][i] / lrow[i];
      o[((size_t)b * SEQ + row) * DIM + h * HEAD_DIM + d] = f2bf(v);
    }
  }
}

// ======================= host launcher =======================
extern "C" void kernel_launch(void* const* d_in, const int* in_sizes, int n_in,
                              void* d_out, int out_size, void* d_ws, size_t ws_size,
                              hipStream_t stream) {
  const float* x      = (const float*)d_in[0];
  const float* ln1_g  = (const float*)d_in[1];
  const float* ln1_b  = (const float*)d_in[2];
  const float* w_qkv  = (const float*)d_in[3];
  const float* w_proj = (const float*)d_in[4];
  const float* b_proj = (const float*)d_in[5];
  const float* ln2_g  = (const float*)d_in[6];
  const float* ln2_b  = (const float*)d_in[7];
  const float* w_fc1  = (const float*)d_in[8];
  const float* b_fc1  = (const float*)d_in[9];
  const float* w_fc2  = (const float*)d_in[10];
  const float* b_fc2  = (const float*)d_in[11];

  // carve workspace
  char* ws = (char*)d_ws;
  size_t off = 0;
  auto carve = [&](size_t bytes) {
    void* p = ws + off;
    off += (bytes + 255) & ~(size_t)255;
    return p;
  };
  bf16_t* wT_qkv  = (bf16_t*)carve((size_t)DIM * 3 * DIM * 2);
  bf16_t* wT_proj = (bf16_t*)carve((size_t)DIM * DIM * 2);
  bf16_t* wT_fc1  = (bf16_t*)carve((size_t)DIM * HIDDEN * 2);
  bf16_t* wT_fc2  = (bf16_t*)carve((size_t)HIDDEN * DIM * 2);
  bf16_t* h1      = (bf16_t*)carve((size_t)NTOK * DIM * 2);
  bf16_t* qkv     = (bf16_t*)carve((size_t)NTOK * 3 * DIM * 2);
  bf16_t* obuf    = (bf16_t*)carve((size_t)NTOK * DIM * 2);
  float*  x1      = (float*)carve((size_t)NTOK * DIM * 4);
  bf16_t* h2      = (bf16_t*)carve((size_t)NTOK * DIM * 2);
  bf16_t* hfc     = (bf16_t*)carve((size_t)NTOK * HIDDEN * 2);
  (void)ws_size; (void)in_sizes; (void)n_in; (void)out_size;

  // weight convert+transpose (f32 [K][N] -> bf16 [N][K])
  auto blocks = [](size_t n) { return (unsigned)((n + 255) / 256); };
  convT_kernel<<<blocks((size_t)DIM * 3 * DIM), 256, 0, stream>>>(w_qkv, wT_qkv, DIM, 3 * DIM);
  convT_kernel<<<blocks((size_t)DIM * DIM), 256, 0, stream>>>(w_proj, wT_proj, DIM, DIM);
  convT_kernel<<<blocks((size_t)DIM * HIDDEN), 256, 0, stream>>>(w_fc1, wT_fc1, DIM, HIDDEN);
  convT_kernel<<<blocks((size_t)HIDDEN * DIM), 256, 0, stream>>>(w_fc2, wT_fc2, HIDDEN, DIM);

  // LN1
  ln_kernel<<<NTOK, 256, 0, stream>>>(x, ln1_g, ln1_b, h1);

  // qkv = h1 @ w_qkv          (bf16 out, no bias)
  gemm_wmma<false, false, false, true>
      <<<dim3(3 * DIM / 64, NTOK / 128), 256, 0, stream>>>(
          h1, wT_qkv, nullptr, nullptr, qkv, NTOK, 3 * DIM, DIM);

  // attention
  attn_kernel<<<dim3(SEQ / 128, HEADS, BATCH), 256, 0, stream>>>(qkv, obuf);

  // x1 = x + o @ w_proj + b_proj   (f32 out, bias + residual)
  gemm_wmma<true, true, false, false>
      <<<dim3(DIM / 64, NTOK / 128), 256, 0, stream>>>(
          obuf, wT_proj, b_proj, x, x1, NTOK, DIM, DIM);

  // LN2
  ln_kernel<<<NTOK, 256, 0, stream>>>(x1, ln2_g, ln2_b, h2);

  // hfc = gelu(h2 @ w_fc1 + b_fc1)  (bf16 out)
  gemm_wmma<true, false, true, true>
      <<<dim3(HIDDEN / 64, NTOK / 128), 256, 0, stream>>>(
          h2, wT_fc1, b_fc1, nullptr, hfc, NTOK, HIDDEN, DIM);

  // out = x1 + hfc @ w_fc2 + b_fc2  (f32 out, bias + residual)
  gemm_wmma<true, true, false, false>
      <<<dim3(DIM / 64, NTOK / 128), 256, 0, stream>>>(
          hfc, wT_fc2, b_fc2, x1, (float*)d_out, NTOK, DIM, HIDDEN);
}
